// RandomAutoEncoder_65386582114521
// MI455X (gfx1250) — matrix operand order repaired and checked by reading, compiled, not run on hardware
//
#include <hip/hip_runtime.h>

// ---------------------------------------------------------------------------
// Problem constants
// ---------------------------------------------------------------------------
#define BB 65536
#define DD 784
#define HH 512
#define ZZ 256

typedef __attribute__((ext_vector_type(16))) __bf16 v16bf;
typedef __attribute__((ext_vector_type(8)))  float  v8f;

struct FragBits { uint4 lo, hi; };   // 32 bytes == v16bf

static __device__ __forceinline__ unsigned short f2bf(float f) {
    // round-to-nearest-even f32 -> bf16 (bits)
    unsigned int u = __float_as_uint(f);
    u += 0x7FFFu + ((u >> 16) & 1u);
    return (unsigned short)(u >> 16);
}

static __device__ __forceinline__ float sigmoidf_(float x) {
    return 1.0f / (1.0f + __expf(-x));
}

static __device__ __forceinline__ float fast_tanh(float x) {
#if __has_builtin(__builtin_amdgcn_tanhf)
    return __builtin_amdgcn_tanhf(x);     // gfx1250 v_tanh_f32
#else
    return tanhf(x);
#endif
}

// low 32 bits of a generic pointer to LDS == LDS byte address (flat aperture rule)
static __device__ __forceinline__ unsigned lds_off(const void* p) {
    return (unsigned)(unsigned long long)p;
}

// gfx1250 async global -> LDS copy, tracked by ASYNCcnt
static __device__ __forceinline__ void async_copy_b128(unsigned ldsaddr, const void* g) {
    asm volatile("global_load_async_to_lds_b128 %0, %1, off"
                 :: "v"(ldsaddr), "v"((unsigned long long)g)
                 : "memory");
}

static __device__ __forceinline__ void wait_async0() {
#if __has_builtin(__builtin_amdgcn_s_wait_asynccnt)
    __builtin_amdgcn_s_wait_asynccnt(0);
#else
    asm volatile("s_wait_asynccnt 0x0" ::: "memory");
#endif
}

#define EPS_F 1.1920929e-7f

// ---------------------------------------------------------------------------
// vectorized f32 -> bf16 conversion (4 elems / thread)
// ---------------------------------------------------------------------------
__global__ __launch_bounds__(256) void cvt4_f32_bf16(const float4* __restrict__ s,
                                                     uint2* __restrict__ d, int n4) {
    int i = blockIdx.x * 256 + threadIdx.x;
    if (i < n4) {
        float4 v = s[i];
        uint2 r;
        r.x = (unsigned)f2bf(v.x) | ((unsigned)f2bf(v.y) << 16);
        r.y = (unsigned)f2bf(v.z) | ((unsigned)f2bf(v.w) << 16);
        d[i] = r;
    }
}

// ---------------------------------------------------------------------------
// Tiled bf16 WMMA GEMM:  Out = act(A @ W + b)
//   A: M x K bf16 (M = 65536), W: K x N row-major bf16, bias: N f32
//   256 thr = 8 waves; block tile 128(M) x 128(N) x 32(K)
//   wave tile 32 x 64 -> 8 WMMA / k-step; double-buffered LDS, 1 barrier/step
//   A staged with global_load_async_to_lds_b128 (ASYNCcnt), B transposed
//   through registers into LDS.
// MODE: 0 = tanh -> bf16 out; 1 = STE hard bits -> f32 (uses U); 3 = sigmoid -> f32
// ---------------------------------------------------------------------------
template <int MODE>
__global__ __launch_bounds__(256) void gemm_mlp(const unsigned short* __restrict__ A,
                                                const unsigned short* __restrict__ Wb,
                                                const float* __restrict__ bias,
                                                void* __restrict__ Outp,
                                                const float* __restrict__ U,
                                                int N, int K) {
    constexpr int LDA = 40;                         // padded row stride (ushorts)
    __shared__ __align__(16) unsigned short As[2][128 * LDA];
    __shared__ __align__(16) unsigned short Bs[2][128 * LDA];

    const int tid  = threadIdx.x;
    const int lane = tid & 31;
    const int wid  = tid >> 5;
    const int wm   = wid >> 1;                      // 0..3 (M)
    const int wn   = wid & 1;                       // 0..1 (N)
    const int half = lane >> 4;                     // 0/1
    const int l15  = lane & 15;

    const long m_base = (long)blockIdx.x * 128;
    const int  n_blk  = blockIdx.y * 128;

    const v8f zero8 = {0.f, 0.f, 0.f, 0.f, 0.f, 0.f, 0.f, 0.f};
    v8f acc[2][4];
#pragma unroll
    for (int i = 0; i < 2; ++i)
#pragma unroll
        for (int j = 0; j < 4; ++j) acc[i][j] = zero8;

    // A tile 128x32: thread handles 8 cols at rows r and r+64
    const int a_row  = tid >> 2;                    // 0..63
    const int a_col8 = (tid & 3) * 8;               // 0,8,16,24
    // B tile 32(K)x128(N): thread handles rows k,k+1 x 8 n's
    const int b_k2   = (tid & 15) * 2;              // 0..30
    const int b_ng8  = (tid >> 4) * 8;              // 0..120
    const int b_n    = n_blk + b_ng8;

    uint4 rB0, rB1;

    auto load_B = [&](int kt) {
        const int kb = kt * 32 + b_k2;
        rB0.x = rB0.y = rB0.z = rB0.w = 0u;
        rB1 = rB0;
        if (b_n < N) {
            if (kb < K)     rB0 = *(const uint4*)(Wb + (long)kb * N + b_n);
            if (kb + 1 < K) rB1 = *(const uint4*)(Wb + (long)(kb + 1) * N + b_n);
        }
    };

    auto store_B = [&](int buf) {
        const unsigned wl[4] = {rB0.x, rB0.y, rB0.z, rB0.w};
        const unsigned wh[4] = {rB1.x, rB1.y, rB1.z, rB1.w};
#pragma unroll
        for (int j = 0; j < 4; ++j) {
            const unsigned p0 = (wl[j] & 0xFFFFu) | (wh[j] << 16);        // n = 2j
            const unsigned p1 = (wl[j] >> 16) | (wh[j] & 0xFFFF0000u);    // n = 2j+1
            *(unsigned*)&Bs[buf][(b_ng8 + 2 * j) * LDA + b_k2]     = p0;
            *(unsigned*)&Bs[buf][(b_ng8 + 2 * j + 1) * LDA + b_k2] = p1;
        }
    };

    auto stage_A = [&](int kt, int buf) {
        const int kc = kt * 32 + a_col8;
        if (kc < K) {
            async_copy_b128(lds_off(&As[buf][a_row * LDA + a_col8]),
                            A + (m_base + a_row) * (long)K + kc);
            async_copy_b128(lds_off(&As[buf][(a_row + 64) * LDA + a_col8]),
                            A + (m_base + a_row + 64) * (long)K + kc);
        } else {                                     // ragged K tail: zero-fill
            uint4 z; z.x = z.y = z.z = z.w = 0u;
            *(uint4*)&As[buf][a_row * LDA + a_col8]        = z;
            *(uint4*)&As[buf][(a_row + 64) * LDA + a_col8] = z;
        }
    };

    const int KT = (K + 31) / 32;

    load_B(0);
    stage_A(0, 0);
    store_B(0);

    for (int kt = 0; kt < KT; ++kt) {
        const int  cur  = kt & 1;
        const bool more = (kt + 1) < KT;
        if (more) load_B(kt + 1);                   // global loads in flight

        wait_async0();                              // A copies into buf[cur] done
        __syncthreads();                            // buf[cur] visible to all

        if (more) stage_A(kt + 1, cur ^ 1);         // async fill of other buffer

        // ---- fragments per ISA 16-bit A/B layouts ----
        v16bf afrag[2], bfrag[4];
#pragma unroll
        for (int i = 0; i < 2; ++i) {
            const int rowL = wm * 32 + i * 16 + l15;
            const unsigned short* p = &As[cur][rowL * LDA + half * 8];
            FragBits f;
            f.lo = *(const uint4*)p;                // K 0..7   (or 8..15)
            f.hi = *(const uint4*)(p + 16);         // K 16..23 (or 24..31)
            afrag[i] = __builtin_bit_cast(v16bf, f);
        }
#pragma unroll
        for (int j = 0; j < 4; ++j) {
            const int nL = wn * 64 + j * 16 + l15;
            const unsigned short* p = &Bs[cur][nL * LDA + half * 8];
            FragBits f;
            f.lo = *(const uint4*)p;                // K 0..7   (or 8..15)
            f.hi = *(const uint4*)(p + 16);         // K 16..23 (or 24..31)
            bfrag[j] = __builtin_bit_cast(v16bf, f);
        }

#pragma unroll
        for (int i = 0; i < 2; ++i)
#pragma unroll
            for (int j = 0; j < 4; ++j)
                acc[i][j] = __builtin_amdgcn_wmma_f32_16x16x32_bf16(
                    false, afrag[i], false, bfrag[j], (short)0, acc[i][j],
                    false, false);

        if (more) store_B(cur ^ 1);                 // B into other buffer
    }

    // ---- epilogue: bias + activation + store ----
#pragma unroll
    for (int j = 0; j < 4; ++j) {
        const int n  = n_blk + wn * 64 + j * 16 + l15;
        const float bv = (n < N) ? bias[n] : 0.f;
#pragma unroll
        for (int i = 0; i < 2; ++i) {
#pragma unroll
            for (int r = 0; r < 8; ++r) {
                const long  m = m_base + wm * 32 + i * 16 + half * 8 + r;
                const float v = acc[i][j][r] + bv;
                if (n < N) {
                    const long idx = m * (long)N + n;
                    if (MODE == 0) {
                        ((unsigned short*)Outp)[idx] = f2bf(fast_tanh(v));
                    } else if (MODE == 1) {
                        float u = U[idx];
                        u = fminf(fmaxf(u, EPS_F), 1.0f - EPS_F);
                        float soft = sigmoidf_(v + __logf(u) - __logf(1.0f - u));
                        soft = fminf(fmaxf(soft, EPS_F), 1.0f - EPS_F);
                        ((float*)Outp)[idx] = rintf(soft);   // hard bits (STE fwd)
                    } else {
                        ((float*)Outp)[idx] = sigmoidf_(v);
                    }
                }
            }
        }
    }
}

// ---------------------------------------------------------------------------
// Random row masking via bitonic selection.
// key = (score_bits << 32) | col reproduces stable argsort order exactly
// (scores are uniform in [0,1) -> positive floats -> bit order == value order).
// mask[j] = key[j] < sorted[k]  <=>  rank(argsort(argsort)) < k.
// One block (256 thr) per row; Z == 256.
// ---------------------------------------------------------------------------
__global__ __launch_bounds__(256) void mask_kernel(const float* __restrict__ scores,
                                                   const float* __restrict__ bern_u,
                                                   const int* __restrict__ kv,
                                                   float* __restrict__ sec1,
                                                   unsigned short* __restrict__ encb) {
    __shared__ unsigned long long s[ZZ];
    const int  row  = blockIdx.x;
    const int  j    = threadIdx.x;
    const long base = (long)row * ZZ;

    const unsigned long long key =
        ((unsigned long long)__float_as_uint(scores[base + j]) << 32) | (unsigned)j;
    s[j] = key;
    __syncthreads();

    // bitonic sort ascending (all keys distinct thanks to index bits)
    for (int k2 = 2; k2 <= ZZ; k2 <<= 1) {
        for (int jj = k2 >> 1; jj > 0; jj >>= 1) {
            const int ixj = j ^ jj;
            if (ixj > j) {
                const unsigned long long a = s[j], b = s[ixj];
                const bool up = ((j & k2) == 0);
                if ((a > b) == up) { s[j] = b; s[ixj] = a; }
            }
            __syncthreads();
        }
    }

    const int kk = kv[row];
    const unsigned long long T = (kk < ZZ) ? s[kk] : ~0ULL;  // kk==Z -> all masked

    float v;
    if (key < T)
        v = (bern_u[base + j] < 0.5f) ? 1.0f : 0.0f;
    else
        v = sec1[base + j];

    sec1[base + j] = v;
    encb[base + j] = f2bf(v);
}

// ---------------------------------------------------------------------------
// Launcher
// ---------------------------------------------------------------------------
extern "C" void kernel_launch(void* const* d_in, const int* in_sizes, int n_in,
                              void* d_out, int out_size, void* d_ws, size_t ws_size,
                              hipStream_t stream) {
    (void)in_sizes; (void)n_in; (void)out_size; (void)ws_size;

    const float* x          = (const float*)d_in[0];
    const float* logistic_u = (const float*)d_in[13];
    const float* scores     = (const float*)d_in[14];
    const float* bern_u     = (const float*)d_in[15];
    const int*   k_vals     = (const int*)d_in[16];

    // workspace carve-out (aligned to 256B)
    char*  ws  = (char*)d_ws;
    size_t off = 0;
    auto take = [&](size_t bytes) -> void* {
        void* r = ws + off;
        off = (off + bytes + 255) & ~(size_t)255;
        return r;
    };

    const size_t wElems[6] = {(size_t)DD * HH, (size_t)HH * ZZ, (size_t)ZZ * ZZ,
                              (size_t)ZZ * ZZ, (size_t)ZZ * HH, (size_t)HH * DD};
    unsigned short* Wb[6];
    for (int i = 0; i < 6; ++i) Wb[i] = (unsigned short*)take(wElems[i] * 2);

    unsigned short* xb   = (unsigned short*)take((size_t)BB * DD * 2); // bf16 x
    unsigned short* bufA = (unsigned short*)take((size_t)BB * HH * 2); // h1 / d5
    unsigned short* bufB = (unsigned short*)take((size_t)BB * ZZ * 2); // h2 / d4
    unsigned short* bufC = xb;   // binary bf16; lifetime disjoint from xb use

    float* sec1 = (float*)d_out;                         // binary_encoded [B,Z]
    float* sec2 = sec1 + (size_t)BB * ZZ;                // decoded        [B,D]

    // 1) f32 -> bf16: weights + x (all sizes divisible by 4)
    for (int i = 0; i < 6; ++i) {
        const int n4 = (int)(wElems[i] / 4);
        cvt4_f32_bf16<<<(n4 + 255) / 256, 256, 0, stream>>>(
            (const float4*)d_in[1 + 2 * i], (uint2*)Wb[i], n4);
    }
    {
        const int n4 = (int)(((size_t)BB * DD) / 4);
        cvt4_f32_bf16<<<(n4 + 255) / 256, 256, 0, stream>>>(
            (const float4*)x, (uint2*)xb, n4);
    }

    const dim3 blk(256);
    auto grid = [](int N) { return dim3(BB / 128, (N + 127) / 128); };

    // 2) encoder
    gemm_mlp<0><<<grid(HH), blk, 0, stream>>>(xb,   Wb[0], (const float*)d_in[2],  bufA, nullptr,    HH, DD);
    gemm_mlp<0><<<grid(ZZ), blk, 0, stream>>>(bufA, Wb[1], (const float*)d_in[4],  bufB, nullptr,    ZZ, HH);
    gemm_mlp<1><<<grid(ZZ), blk, 0, stream>>>(bufB, Wb[2], (const float*)d_in[6],  sec1, logistic_u, ZZ, ZZ);

    // 3) random row masking -> final binary (f32 in d_out, bf16 for decoder)
    mask_kernel<<<BB, blk, 0, stream>>>(scores, bern_u, k_vals, sec1, bufC);

    // 4) decoder
    gemm_mlp<0><<<grid(ZZ), blk, 0, stream>>>(bufC, Wb[3], (const float*)d_in[8],  bufB, nullptr, ZZ, ZZ);
    gemm_mlp<0><<<grid(HH), blk, 0, stream>>>(bufB, Wb[4], (const float*)d_in[10], bufA, nullptr, HH, ZZ);
    gemm_mlp<3><<<grid(DD), blk, 0, stream>>>(bufA, Wb[5], (const float*)d_in[12], sec2, nullptr, DD, HH);
}
